// MultiHeadAttention_34110630265011
// MI455X (gfx1250) — matrix operand order repaired and checked by reading, not compiled, not run on hardware
//
#include <hip/hip_runtime.h>
#include <hip/hip_bf16.h>

typedef __bf16 bf16_t;
typedef __attribute__((ext_vector_type(16))) __bf16 v16bf;
typedef __attribute__((ext_vector_type(8)))  __bf16 v8bf;
typedef __attribute__((ext_vector_type(8)))  float   v8f;

#define DEV __device__ __forceinline__

constexpr int BB = 4, T = 2048, D = 1024, H = 16, HD = 64;
constexpr int BT = BB * T;            // 8192 rows
constexpr int WAVES_PER_BLOCK = 8;    // 256 threads, wave32

// ---------------------------------------------------------------------------
// WMMA wrapper: D = A(16x32 bf16) * B(32x16 bf16) + C(f32)
// ---------------------------------------------------------------------------
DEV v8f wmma_bf16(v16bf a, v16bf b, v8f c) {
  return __builtin_amdgcn_wmma_f32_16x16x32_bf16(
      /*neg_a=*/false, a, /*neg_b=*/false, b,
      /*c_mod=*/(short)0, c, /*reuse_a=*/false, /*reuse_b=*/false);
}

DEV void wait_lds() {
#if defined(__gfx1250__)
  asm volatile("s_wait_dscnt 0" ::: "memory");
#endif
}

// ---------------------------------------------------------------------------
// Fragment loaders, CDNA5 ISA 7.12.2 VGPR layouts (wave32).
// A (16x32, 16-bit): lane L -> M = L&15; halves [0..7] = K(k0 + hi*8 ..),
//                    halves [8..15] = K(k0 + 16 + hi*8 ..), hi = L>=16.
// ---------------------------------------------------------------------------
DEV v16bf load_a(const bf16_t* __restrict__ src, int ld, int m0, int k0, int lane) {
  const bf16_t* row = src + (size_t)(m0 + (lane & 15)) * ld;
  int koff = ((lane >> 4) & 1) * 8;
  union { v16bf v; v8bf h[2]; } u;
  u.h[0] = *(const v8bf*)(row + k0 + koff);
  u.h[1] = *(const v8bf*)(row + k0 + 16 + koff);
  return u.v;
}

// B (32x16, 16-bit), source stored row-major [N][K] (K contiguous per n-row):
// lane L -> N = L&15; halves [0..7] = K(k0 + hi*8 ..), halves [8..15] = K(k0 + 16 + hi*8 ..).
DEV v16bf load_b_nt(const bf16_t* __restrict__ src, int ld, int n0, int k0, int lane) {
  const bf16_t* row = src + (size_t)(n0 + (lane & 15)) * ld;
  int kb = k0 + ((lane >> 4) & 1) * 8;
  union { v16bf v; v8bf h[2]; } u;
  u.h[0] = *(const v8bf*)(row + kb);
  u.h[1] = *(const v8bf*)(row + kb + 16);
  return u.v;
}

// ---------------------------------------------------------------------------
// fp32 -> bf16 conversion
// ---------------------------------------------------------------------------
__global__ void cvt_bf16(const float* __restrict__ s, bf16_t* __restrict__ d, int n) {
  int i = blockIdx.x * blockDim.x + threadIdx.x;
  if (i < n) d[i] = (bf16_t)s[i];
}

// ---------------------------------------------------------------------------
// Fused QKV projection: out = x @ W^T (NT GEMM). Wave tile = 32(M) x 64(N):
// 2 A-frags x 4 B-frags -> 8 WMMAs per 32-wide k-step, 8 resident C-frags.
// Q scaled by 1/sqrt(HD), written [B][H][T][HD]. K written [B][H][T][HD].
// V written TRANSPOSED [B][H][HD][T] so attention's P*V B-fragments are
// contiguous loads.
// ---------------------------------------------------------------------------
__global__ __launch_bounds__(256) void qkv_gemm(
    const bf16_t* __restrict__ xb,
    const bf16_t* __restrict__ wq, const bf16_t* __restrict__ wk,
    const bf16_t* __restrict__ wv,
    bf16_t* __restrict__ qo, bf16_t* __restrict__ ko, bf16_t* __restrict__ vt) {
  const int lane = threadIdx.x & 31;
  const int wave = threadIdx.x >> 5;
  int tile = blockIdx.x * WAVES_PER_BLOCK + wave;
  const int perMat = (BT / 32) * (D / 64);          // 256 * 16 = 4096
  int mat = tile / perMat;
  int rem = tile - mat * perMat;
  int m0 = (rem >> 4) * 32;                         // 0..8160
  int n0 = (rem & 15) * 64;                         // 0..960
  const bf16_t* W = (mat == 0) ? wq : (mat == 1) ? wk : wv;

  v8f acc[2][4] = {};
  for (int k = 0; k < D; k += 32) {
    v16bf a0 = load_a(xb, D, m0,      k, lane);
    v16bf a1 = load_a(xb, D, m0 + 16, k, lane);
    v16bf b0 = load_b_nt(W, D, n0,      k, lane);
    v16bf b1 = load_b_nt(W, D, n0 + 16, k, lane);
    v16bf b2 = load_b_nt(W, D, n0 + 32, k, lane);
    v16bf b3 = load_b_nt(W, D, n0 + 48, k, lane);
    acc[0][0] = wmma_bf16(a0, b0, acc[0][0]);
    acc[0][1] = wmma_bf16(a0, b1, acc[0][1]);
    acc[0][2] = wmma_bf16(a0, b2, acc[0][2]);
    acc[0][3] = wmma_bf16(a0, b3, acc[0][3]);
    acc[1][0] = wmma_bf16(a1, b0, acc[1][0]);
    acc[1][1] = wmma_bf16(a1, b1, acc[1][1]);
    acc[1][2] = wmma_bf16(a1, b2, acc[1][2]);
    acc[1][3] = wmma_bf16(a1, b3, acc[1][3]);
  }

  const float scale = (mat == 0) ? 0.125f : 1.0f;   // 1/sqrt(64) folded into Q
  const int hi = lane >> 4;
  const int ln = lane & 15;
  const int h  = n0 >> 6;                           // whole 64-wide N tile = one head
#pragma unroll
  for (int i = 0; i < 2; ++i) {
#pragma unroll
    for (int j = 0; j < 4; ++j) {
      const int hd = j * 16 + ln;
#pragma unroll
      for (int r = 0; r < 8; ++r) {
        int m  = m0 + i * 16 + r + 8 * hi;
        int b_ = m >> 11, t = m & (T - 1);
        float val = acc[i][j][r] * scale;
        if (mat == 2) {
          vt[(((size_t)b_ * H + h) * HD + hd) * T + t] = (bf16_t)val;
        } else {
          bf16_t* dst = (mat == 0) ? qo : ko;
          dst[(((size_t)b_ * H + h) * T + t) * HD + hd] = (bf16_t)val;
        }
      }
    }
  }
}

// ---------------------------------------------------------------------------
// Flash attention: one wave owns 16 query rows of one (b,h); streams 32-wide
// causal k-tiles with online softmax. 8 WMMAs per k-tile (4 for S, 4 for P*V).
// P is re-laid C-layout -> A-layout via a per-wave LDS tile.
// V is pre-transposed [bh][HD][T] so its B-fragments load contiguously.
// ---------------------------------------------------------------------------
__global__ __launch_bounds__(256) void attn_fwd(
    const bf16_t* __restrict__ qb, const bf16_t* __restrict__ kb,
    const bf16_t* __restrict__ vt, bf16_t* __restrict__ ao) {
  __shared__ __align__(16) bf16_t pTile[WAVES_PER_BLOCK][16 * 32];
  const int lane = threadIdx.x & 31;
  const int wave = threadIdx.x >> 5;
  int gid = blockIdx.x * WAVES_PER_BLOCK + wave;
  int bh = gid >> 7;                 // T/16 = 128 q-tiles per head
  int qt = gid & 127;
  const bf16_t* Qp = qb + (size_t)bh * T * HD;   // [T][HD]
  const bf16_t* Kp = kb + (size_t)bh * T * HD;   // [T][HD]
  const bf16_t* Vp = vt + (size_t)bh * HD * T;   // [HD][T] (transposed)
  bf16_t* pt = pTile[wave];

  const int hi = lane >> 4;
  const int ln = lane & 15;
  const int qmax = qt * 16 + 15;

  // Q fragments for K(dim)=64: two A fragments, kept resident.
  v16bf aq0 = load_a(Qp, HD, qt * 16, 0, lane);
  v16bf aq1 = load_a(Qp, HD, qt * 16, 32, lane);

  v8f o0 = {}, o1 = {}, o2 = {}, o3 = {};
  float rm[8], rl[8];
#pragma unroll
  for (int r = 0; r < 8; ++r) { rm[r] = -1e30f; rl[r] = 0.0f; }

  for (int kt = 0; kt <= qmax; kt += 32) {
    // S = Q * K^T for 32 k-positions (two 16x16 fragments)
    v8f s0 = {}, s1 = {};
    {
      v16bf b0 = load_b_nt(Kp, HD, kt, 0, lane);
      v16bf b1 = load_b_nt(Kp, HD, kt, 32, lane);
      s0 = wmma_bf16(aq0, b0, s0);
      s0 = wmma_bf16(aq1, b1, s0);
      v16bf b2 = load_b_nt(Kp, HD, kt + 16, 0, lane);
      v16bf b3 = load_b_nt(Kp, HD, kt + 16, 32, lane);
      s1 = wmma_bf16(aq0, b2, s1);
      s1 = wmma_bf16(aq1, b3, s1);
    }
    const int kp0 = kt + ln, kp1 = kt + 16 + ln;

    // causal mask + tile row-max
    float tm[8];
#pragma unroll
    for (int r = 0; r < 8; ++r) {
      int qp = qt * 16 + r + 8 * hi;
      if (kp0 > qp) s0[r] = -1e30f;
      if (kp1 > qp) s1[r] = -1e30f;
      tm[r] = fmaxf(s0[r], s1[r]);
    }
#pragma unroll
    for (int off = 1; off < 16; off <<= 1)
#pragma unroll
      for (int r = 0; r < 8; ++r)
        tm[r] = fmaxf(tm[r], __shfl_xor(tm[r], off, 32));

    // online softmax update
    float al[8], ps[8];
#pragma unroll
    for (int r = 0; r < 8; ++r) {
      float nm = fmaxf(rm[r], tm[r]);
      al[r] = __expf(rm[r] - nm);
      rm[r] = nm;
      int qp = qt * 16 + r + 8 * hi;
      float p0 = (kp0 > qp) ? 0.0f : __expf(s0[r] - nm);
      float p1 = (kp1 > qp) ? 0.0f : __expf(s1[r] - nm);
      s0[r] = p0; s1[r] = p1;
      ps[r] = p0 + p1;
    }
#pragma unroll
    for (int off = 1; off < 16; off <<= 1)
#pragma unroll
      for (int r = 0; r < 8; ++r)
        ps[r] += __shfl_xor(ps[r], off, 32);
#pragma unroll
    for (int r = 0; r < 8; ++r) {
      rl[r] = al[r] * rl[r] + ps[r];
      o0[r] *= al[r]; o1[r] *= al[r]; o2[r] *= al[r]; o3[r] *= al[r];
    }

    // P: C-layout -> LDS -> A-layout (wave-private tile; DS ops in-order)
#pragma unroll
    for (int r = 0; r < 8; ++r) {
      int m = r + 8 * hi;
      pt[m * 32 + ln]      = (bf16_t)s0[r];
      pt[m * 32 + 16 + ln] = (bf16_t)s1[r];
    }
    wait_lds();
    v16bf aP = load_a(pt, 32, 0, 0, lane);
    wait_lds();

    // O += P * V. V^T rows are hd with t contiguous -> contiguous B loads.
    v16bf bv0 = load_b_nt(Vp, T, 0,  kt, lane);
    v16bf bv1 = load_b_nt(Vp, T, 16, kt, lane);
    v16bf bv2 = load_b_nt(Vp, T, 32, kt, lane);
    v16bf bv3 = load_b_nt(Vp, T, 48, kt, lane);
    o0 = wmma_bf16(aP, bv0, o0);
    o1 = wmma_bf16(aP, bv1, o1);
    o2 = wmma_bf16(aP, bv2, o2);
    o3 = wmma_bf16(aP, bv3, o3);
  }

  // normalize and write back as [B][T][D] bf16 (input to output projection)
  int b_ = bh >> 4, h = bh & (H - 1);
#pragma unroll
  for (int r = 0; r < 8; ++r) {
    float inv = 1.0f / rl[r];
    int t = qt * 16 + r + 8 * hi;
    size_t row = ((size_t)b_ * T + t) * D + h * HD;
    ao[row + ln]      = (bf16_t)(o0[r] * inv);
    ao[row + 16 + ln] = (bf16_t)(o1[r] * inv);
    ao[row + 32 + ln] = (bf16_t)(o2[r] * inv);
    ao[row + 48 + ln] = (bf16_t)(o3[r] * inv);
  }
}

// ---------------------------------------------------------------------------
// Output projection: out = attn_out @ Wp^T + bp, fp32 result.
// Wave tile = 32(M) x 64(N), same structure as qkv_gemm.
// ---------------------------------------------------------------------------
__global__ __launch_bounds__(256) void proj_gemm(
    const bf16_t* __restrict__ ab, const bf16_t* __restrict__ wp,
    const float* __restrict__ bp, float* __restrict__ out) {
  const int lane = threadIdx.x & 31;
  const int wave = threadIdx.x >> 5;
  int tile = blockIdx.x * WAVES_PER_BLOCK + wave;
  int m0 = (tile >> 4) * 32;
  int n0 = (tile & 15) * 64;

  v8f acc[2][4] = {};
  for (int k = 0; k < D; k += 32) {
    v16bf a0 = load_a(ab, D, m0,      k, lane);
    v16bf a1 = load_a(ab, D, m0 + 16, k, lane);
    v16bf b0 = load_b_nt(wp, D, n0,      k, lane);
    v16bf b1 = load_b_nt(wp, D, n0 + 16, k, lane);
    v16bf b2 = load_b_nt(wp, D, n0 + 32, k, lane);
    v16bf b3 = load_b_nt(wp, D, n0 + 48, k, lane);
    acc[0][0] = wmma_bf16(a0, b0, acc[0][0]);
    acc[0][1] = wmma_bf16(a0, b1, acc[0][1]);
    acc[0][2] = wmma_bf16(a0, b2, acc[0][2]);
    acc[0][3] = wmma_bf16(a0, b3, acc[0][3]);
    acc[1][0] = wmma_bf16(a1, b0, acc[1][0]);
    acc[1][1] = wmma_bf16(a1, b1, acc[1][1]);
    acc[1][2] = wmma_bf16(a1, b2, acc[1][2]);
    acc[1][3] = wmma_bf16(a1, b3, acc[1][3]);
  }

  const int hi = lane >> 4;
  const int ln = lane & 15;
#pragma unroll
  for (int i = 0; i < 2; ++i) {
#pragma unroll
    for (int j = 0; j < 4; ++j) {
      const int nn = n0 + j * 16 + ln;
      const float bias = bp[nn];
#pragma unroll
      for (int r = 0; r < 8; ++r) {
        int m = m0 + i * 16 + r + 8 * hi;
        out[(size_t)m * D + nn] = acc[i][j][r] + bias;
      }
    }
  }
}

// ---------------------------------------------------------------------------
// Host launch. Input order (setup_inputs): x, Wk, Wq, Wv, Wp, bp.
// ---------------------------------------------------------------------------
extern "C" void kernel_launch(void* const* d_in, const int* in_sizes, int n_in,
                              void* d_out, int out_size, void* d_ws, size_t ws_size,
                              hipStream_t stream) {
  (void)in_sizes; (void)n_in; (void)out_size; (void)ws_size;
  const float* x  = (const float*)d_in[0];
  const float* Wk = (const float*)d_in[1];
  const float* Wq = (const float*)d_in[2];
  const float* Wv = (const float*)d_in[3];
  const float* Wp = (const float*)d_in[4];
  const float* bp = (const float*)d_in[5];
  float* out = (float*)d_out;

  bf16_t* ws = (bf16_t*)d_ws;
  size_t off = 0;
  bf16_t* xb  = ws + off; off += (size_t)BT * D;   // x bf16
  bf16_t* wqb = ws + off; off += (size_t)D * D;
  bf16_t* wkb = ws + off; off += (size_t)D * D;
  bf16_t* wvb = ws + off; off += (size_t)D * D;
  bf16_t* wpb = ws + off; off += (size_t)D * D;
  bf16_t* qb  = ws + off; off += (size_t)BT * D;   // [B][H][T][HD]
  bf16_t* kb  = ws + off; off += (size_t)BT * D;   // [B][H][T][HD]
  bf16_t* vtb = ws + off; off += (size_t)BT * D;   // [B][H][HD][T] (transposed)
  bf16_t* ab  = ws + off; off += (size_t)BT * D;   // attn out, [B][T][D]

  const int nx = BT * D;   // 8,388,608
  const int nw = D * D;    // 1,048,576
  cvt_bf16<<<(nx + 255) / 256, 256, 0, stream>>>(x,  xb,  nx);
  cvt_bf16<<<(nw + 255) / 256, 256, 0, stream>>>(Wq, wqb, nw);
  cvt_bf16<<<(nw + 255) / 256, 256, 0, stream>>>(Wk, wkb, nw);
  cvt_bf16<<<(nw + 255) / 256, 256, 0, stream>>>(Wv, wvb, nw);
  cvt_bf16<<<(nw + 255) / 256, 256, 0, stream>>>(Wp, wpb, nw);

  // 3 mats * 256 mTiles * 16 nTiles = 12288 wave-tiles / 8 waves = 1536 blocks
  qkv_gemm<<<1536, 256, 0, stream>>>(xb, wqb, wkb, wvb, qb, kb, vtb);

  // 64 (b*h) * 128 q-tiles = 8192 waves / 8 = 1024 blocks
  attn_fwd<<<1024, 256, 0, stream>>>(qb, kb, vtb, ab);

  // 256 * 16 = 4096 wave-tiles / 8 = 512 blocks
  proj_gemm<<<512, 256, 0, stream>>>(ab, wpb, bp, out);
}
